// FocusedAttnV1_65859028517417
// MI455X (gfx1250) — compile-verified, matrix operand order checked by reading, not run on hardware
//
#include <hip/hip_runtime.h>

// Fused windowed attention for MI455X (gfx1250, wave32, WMMA f16 -> f32 acc).
// One workgroup (8 wave32s) per batch element. All intermediates in LDS.
// Weights are repacked once per launch into d_ws as f16 fragments in the exact
// wave32 WMMA B-operand layout, so each B load is one contiguous v16h.
// Stage-1 K-loops are FULLY unrolled so the A-fragment register cache
// (areg[12], 96 VGPRs) is never dynamically indexed -> no scratch spills.

typedef __attribute__((ext_vector_type(16))) _Float16 v16h;
typedef __attribute__((ext_vector_type(8)))  float    v8f;

constexpr int kHeads = 6;
constexpr int kDh    = 64;
constexpr int kC     = 384;   // = kHeads * kDh
constexpr int kNk    = 64;
constexpr int kNq    = 27;

constexpr int    kFragHalves = 32 * 16;                       // 512 f16 per fragment
constexpr int    kFragsPerW  = 12 * 24;                       // (K/32) x (N/16) = 288
constexpr size_t kWsHalvesPerW = (size_t)kFragsPerW * kFragHalves;
constexpr size_t kWsNeededBytes = 3 * kWsHalvesPerW * sizeof(_Float16);  // 884736 B

static __device__ __forceinline__ v8f wmma16(v16h a, v16h b, v8f c) {
  // D = A(16x32 f16) * B(32x16 f16) + C(16x16 f32)
  return __builtin_amdgcn_wmma_f32_16x16x32_f16(
      /*neg_a=*/false, a, /*neg_b=*/false, b,
      /*c_mod=*/(short)0, c, /*reuse_a=*/false, /*reuse_b=*/false);
}

// --- Fragment loaders (wave32 layouts per CDNA5 ISA 7.12.2) -----------------
// A (16x32, MxK): lane&15 = row M; lane>>4 selects K-subrange.
//   elems 0..7  -> K = kb+0..7,  elems 8..15 -> K = 16+kb+0..7, kb = (lane>>4)*8
// B (32x16, KxN): lane&15 = col N; same K distribution across elements.
// C/D (16x16 f32): vgpr r, lanes 0-15 -> M=r; lanes 16-31 -> M=8+r; col = lane&15.

static __device__ __forceinline__ v16h load_a_f32(const float* A, int ld, int m0,
                                                  int k0, int lane, int mMax) {
  v16h a;
  const int m  = m0 + (lane & 15);
  const int kb = (lane >> 4) * 8;
  if (m < mMax) {
    const float* p = A + (size_t)m * ld + k0 + kb;
#pragma unroll
    for (int j = 0; j < 8; ++j) { a[j] = (_Float16)p[j]; a[8 + j] = (_Float16)p[16 + j]; }
  } else {
#pragma unroll
    for (int j = 0; j < 16; ++j) a[j] = (_Float16)0.0f;
  }
  return a;
}

static __device__ __forceinline__ v16h load_b_f32(const float* B, int ld, int k0,
                                                  int n0, int lane) {
  v16h b;
  const int n  = n0 + (lane & 15);
  const int kb = (lane >> 4) * 8;
  const float* p = B + (size_t)(k0 + kb) * ld + n;
#pragma unroll
  for (int j = 0; j < 8; ++j) {
    b[j]     = (_Float16)p[(size_t)j * ld];
    b[8 + j] = (_Float16)p[(size_t)(16 + j) * ld];
  }
  return b;
}

// Packed f16 weight fragment: one contiguous 32B-per-lane load (2x b128).
static __device__ __forceinline__ v16h load_b_packed(const _Float16* wsW, int kc,
                                                     int nt, int lane) {
  return *reinterpret_cast<const v16h*>(
      wsW + ((size_t)(kc * 24 + nt) * 32 + lane) * 16);
}

static __device__ __forceinline__ v16h load_a_lds(const _Float16* A, int ld, int m0,
                                                  int k0, int lane) {
  v16h a;
  const int m  = m0 + (lane & 15);
  const int kb = (lane >> 4) * 8;
  const _Float16* p = A + (size_t)m * ld + k0 + kb;
#pragma unroll
  for (int j = 0; j < 8; ++j) { a[j] = p[j]; a[8 + j] = p[16 + j]; }
  return a;
}

static __device__ __forceinline__ v16h load_b_lds(const _Float16* B, int ld, int k0,
                                                  int n0, int lane) {
  v16h b;
  const int n  = n0 + (lane & 15);
  const int kb = (lane >> 4) * 8;
  const _Float16* p = B + (size_t)(k0 + kb) * ld + n;
#pragma unroll
  for (int j = 0; j < 8; ++j) { b[j] = p[j * ld]; b[8 + j] = p[(16 + j) * ld]; }
  return b;
}

// Logical B[k][n] = M[n][k]  (used for Q @ K^T without materializing K^T)
static __device__ __forceinline__ v16h load_bT_lds(const _Float16* M, int ld, int k0,
                                                   int n0, int lane) {
  v16h b;
  const int n  = n0 + (lane & 15);
  const int kb = (lane >> 4) * 8;
  const _Float16* p = M + (size_t)n * ld + k0 + kb;
#pragma unroll
  for (int j = 0; j < 8; ++j) { b[j] = p[j]; b[8 + j] = p[16 + j]; }
  return b;
}

// ---- Weight repack: f32 row-major -> f16 fragment-ordered (one wave/frag) ---
__global__ __launch_bounds__(32)
void repack_weights_kernel(const float* __restrict__ Wk,
                           const float* __restrict__ Wv,
                           const float* __restrict__ Wp,
                           _Float16* __restrict__ ws) {
  const int f  = blockIdx.x;            // 0..863
  const int w  = f / kFragsPerW;
  const int fr = f % kFragsPerW;
  const int kc = fr / 24, nt = fr % 24;
  const float* W = (w == 0) ? Wk : (w == 1) ? Wv : Wp;
  const int lane = threadIdx.x;
  const int n  = nt * 16 + (lane & 15);
  const int kb = (lane >> 4) * 8;
  _Float16* dst = ws + ((size_t)f * 32 + lane) * 16;
  const float* p = W + (size_t)(kc * 32 + kb) * kC + n;
#pragma unroll
  for (int j = 0; j < 8; ++j) {
    dst[j]     = (_Float16)p[(size_t)j * kC];
    dst[8 + j] = (_Float16)p[(size_t)(16 + j) * kC];
  }
}

template <bool PACKED>
__global__ __launch_bounds__(256)
void focused_attn_kernel(const float* __restrict__ src_k,
                         const float* __restrict__ src_q,
                         const float* __restrict__ Wk, const float* __restrict__ bk,
                         const float* __restrict__ Wv, const float* __restrict__ bv,
                         const float* __restrict__ pos_bias,
                         const float* __restrict__ Wp, const float* __restrict__ bp,
                         const _Float16* __restrict__ wsK,
                         const _Float16* __restrict__ wsV,
                         const _Float16* __restrict__ wsP,
                         float* __restrict__ out) {
  // 144 KB LDS: Kp(64x384) | Vp(64x384) | Qp(32x384) | P(6x32x64); X aliases Kp.
  __shared__ _Float16 smem[73728];
  _Float16* sKp = smem;            // 24576 halves
  _Float16* sVp = smem + 24576;    // 24576
  _Float16* sQp = smem + 49152;    // 12288
  _Float16* sP  = smem + 61440;    // 12288
  _Float16* sX  = smem;            // 12288 (aliases sKp, used after barrier)

  const int b     = blockIdx.x;
  const int lane  = threadIdx.x & 31;
  const int wave  = threadIdx.x >> 5;
  const int lidx  = lane & 15;
  const int lhalf = lane >> 4;

  const float* srcKb = src_k + (size_t)b * kNk * kC;
  const float* srcQb = src_q + (size_t)b * kNq * kC;

  // ---------------- Stage 1: K/V/Q projections -> LDS (f16) ----------------
  // 12 units over 8 waves (2 passes): units 0..7 = K+V half-strips
  // (mt = u>>1, 12 N-tiles), units 8..11 = Q half-strips. Each unit loads its
  // A strip (16 rows x 384) into registers ONCE and reuses it for all N-tiles
  // (and for both Wk and Wv in the K/V units). K-loops fully unrolled so the
  // areg[] indices are constants (registers, not scratch).
  for (int pass = 0; pass < 2; ++pass) {
    const int u = wave + pass * 8;
    if (u < 8) {
      const int mt = u >> 1, ntb = (u & 1) * 12;
      v16h areg[12];
#pragma unroll
      for (int kc = 0; kc < 12; ++kc)
        areg[kc] = load_a_f32(srcKb, kC, mt * 16, kc * 32, lane, kNk);
      for (int nn = 0; nn < 12; ++nn) {
        const int nt = ntb + nn;
        v8f accK = {}, accV = {};
#pragma unroll
        for (int kc = 0; kc < 12; ++kc) {
          v16h bK = PACKED ? load_b_packed(wsK, kc, nt, lane)
                           : load_b_f32(Wk, kC, kc * 32, nt * 16, lane);
          accK = wmma16(areg[kc], bK, accK);
          v16h bV = PACKED ? load_b_packed(wsV, kc, nt, lane)
                           : load_b_f32(Wv, kC, kc * 32, nt * 16, lane);
          accV = wmma16(areg[kc], bV, accV);
        }
        const float bbK = bk[nt * 16 + lidx];
        const float bbV = bv[nt * 16 + lidx];
        const int m0 = mt * 16 + lhalf * 8;
#pragma unroll
        for (int r = 0; r < 8; ++r) {
          sKp[(size_t)(m0 + r) * kC + nt * 16 + lidx] = (_Float16)(accK[r] + bbK);
          sVp[(size_t)(m0 + r) * kC + nt * 16 + lidx] = (_Float16)(accV[r] + bbV);
        }
      }
    } else if (u < 12) {
      const int uu = u - 8;
      const int mt = uu >> 1, ntb = (uu & 1) * 12;
      v16h areg[12];
#pragma unroll
      for (int kc = 0; kc < 12; ++kc)
        areg[kc] = load_a_f32(srcQb, kC, mt * 16, kc * 32, lane, kNq);  // zero-pad
      for (int nn = 0; nn < 12; ++nn) {
        const int nt = ntb + nn;
        v8f acc = {};
#pragma unroll
        for (int kc = 0; kc < 12; ++kc) {
          v16h bK = PACKED ? load_b_packed(wsK, kc, nt, lane)
                           : load_b_f32(Wk, kC, kc * 32, nt * 16, lane);
          acc = wmma16(areg[kc], bK, acc);  // q uses k_proj (ref quirk)
        }
        const float bb = bk[nt * 16 + lidx];
        const int m0 = mt * 16 + lhalf * 8;
#pragma unroll
        for (int r = 0; r < 8; ++r)
          sQp[(size_t)(m0 + r) * kC + nt * 16 + lidx] =
              (_Float16)((acc[r] + bb) * 0.125f);  // Dh^-0.5
      }
    }
  }
  __syncthreads();

  // -------- Stage 2a: S = Q K^T + pos_bias, softmax over Nk=64, -> P --------
  for (int i = 0; i < 2; ++i) {
    const int u = wave * 2 + i;
    if (u < 12) {
      const int h = u >> 1, mt = u & 1;
      v8f s[4] = {v8f{}, v8f{}, v8f{}, v8f{}};
#pragma unroll
      for (int nt = 0; nt < 4; ++nt)
#pragma unroll
        for (int kc = 0; kc < 2; ++kc) {
          v16h af = load_a_lds(sQp, kC, mt * 16, h * kDh + kc * 32, lane);
          v16h bf = load_bT_lds(sKp, kC, h * kDh + kc * 32, nt * 16, lane);
          s[nt] = wmma16(af, bf, s[nt]);
        }
      // Row m = mt*16 + lhalf*8 + r lives in 16 lanes of one half-wave across
      // the 4 accumulators -> shfl_xor 1/2/4/8 stays inside the half.
#pragma unroll
      for (int r = 0; r < 8; ++r) {
        const int q  = mt * 16 + lhalf * 8 + r;
        const int qq = q < kNq ? q : kNq - 1;  // clamp pad rows (no OOB)
        float v[4];
#pragma unroll
        for (int nt = 0; nt < 4; ++nt)
          v[nt] = s[nt][r] +
                  pos_bias[((size_t)h * kNq + qq) * kNk + nt * 16 + lidx];
        float mx = fmaxf(fmaxf(v[0], v[1]), fmaxf(v[2], v[3]));
        for (int off = 1; off < 16; off <<= 1)
          mx = fmaxf(mx, __shfl_xor(mx, off, 32));
        float e[4], sum = 0.0f;
#pragma unroll
        for (int nt = 0; nt < 4; ++nt) { e[nt] = __expf(v[nt] - mx); sum += e[nt]; }
        for (int off = 1; off < 16; off <<= 1) sum += __shfl_xor(sum, off, 32);
        const float inv = 1.0f / sum;
#pragma unroll
        for (int nt = 0; nt < 4; ++nt) s[nt][r] = e[nt] * inv;
      }
      _Float16* Ph = sP + h * 32 * 64;
      const int m0 = mt * 16 + lhalf * 8;
#pragma unroll
      for (int nt = 0; nt < 4; ++nt) {
        const int n = nt * 16 + lidx;
#pragma unroll
        for (int r = 0; r < 8; ++r)
          Ph[(size_t)(m0 + r) * 64 + n] = (_Float16)s[nt][r];
      }
    }
  }
  __syncthreads();  // P written; all Kp reads done -> sX may alias sKp

  // ---------------- Stage 2b: X = P @ V  (per head) -> LDS -----------------
  for (int i = 0; i < 2; ++i) {
    const int u = wave * 2 + i;
    if (u < 12) {
      const int h = u >> 1, mt = u & 1;
      const _Float16* Ph = sP + h * 32 * 64;
#pragma unroll
      for (int nt = 0; nt < 4; ++nt) {
        v8f x = {};
#pragma unroll
        for (int kc = 0; kc < 2; ++kc) {
          v16h af = load_a_lds(Ph, 64, mt * 16, kc * 32, lane);
          v16h bf = load_b_lds(sVp, kC, kc * 32, h * kDh + nt * 16, lane);
          x = wmma16(af, bf, x);
        }
        const int n  = h * kDh + nt * 16 + lidx;
        const int m0 = mt * 16 + lhalf * 8;
#pragma unroll
        for (int r = 0; r < 8; ++r)
          sX[(size_t)(m0 + r) * kC + n] = (_Float16)x[r];
      }
    }
  }
  __syncthreads();

  // ---------------- Stage 3: out = X @ Wp + bp (rows < 27) -----------------
  float* outb = out + (size_t)b * kNq * kC;
  for (int t = wave; t < 48; t += 8) {
    const int mt = t / 24, nt = t % 24;
    v8f acc = {};
#pragma unroll 4
    for (int kc = 0; kc < 12; ++kc) {
      v16h af = load_a_lds(sX, kC, mt * 16, kc * 32, lane);
      v16h bf = PACKED ? load_b_packed(wsP, kc, nt, lane)
                       : load_b_f32(Wp, kC, kc * 32, nt * 16, lane);
      acc = wmma16(af, bf, acc);
    }
    const float bb = bp[nt * 16 + lidx];
    const int m0 = mt * 16 + lhalf * 8;
#pragma unroll
    for (int r = 0; r < 8; ++r) {
      const int m = m0 + r;
      if (m < kNq) outb[(size_t)m * kC + nt * 16 + lidx] = acc[r] + bb;
    }
  }
}

extern "C" void kernel_launch(void* const* d_in, const int* in_sizes, int n_in,
                              void* d_out, int out_size, void* d_ws, size_t ws_size,
                              hipStream_t stream) {
  (void)n_in; (void)out_size;
  const float* src_k    = (const float*)d_in[0];
  const float* src_q    = (const float*)d_in[1];
  const float* Wk       = (const float*)d_in[2];
  const float* bk       = (const float*)d_in[3];
  const float* Wv       = (const float*)d_in[4];
  const float* bv       = (const float*)d_in[5];
  const float* pos_bias = (const float*)d_in[6];
  const float* Wp       = (const float*)d_in[7];
  const float* bp       = (const float*)d_in[8];
  float* out            = (float*)d_out;

  const int B = in_sizes[0] / (kNk * kC);  // 4096

  const bool packed = (d_ws != nullptr) && (ws_size >= kWsNeededBytes);
  if (packed) {
    _Float16* ws = (_Float16*)d_ws;
    repack_weights_kernel<<<3 * kFragsPerW, 32, 0, stream>>>(Wk, Wv, Wp, ws);
    focused_attn_kernel<true><<<B, 256, 0, stream>>>(
        src_k, src_q, Wk, bk, Wv, bv, pos_bias, Wp, bp,
        ws, ws + kWsHalvesPerW, ws + 2 * kWsHalvesPerW, out);
  } else {
    focused_attn_kernel<false><<<B, 256, 0, stream>>>(
        src_k, src_q, Wk, bk, Wv, bv, pos_bias, Wp, bp,
        nullptr, nullptr, nullptr, out);
  }
}
